// Layer_14714557956495
// MI455X (gfx1250) — compile-verified, matrix operand order checked by reading, not run on hardware
//
#include <hip/hip_runtime.h>

// ---------------------------------------------------------------- constants
#define BATCH   256
#define SEQ     200
#define DMODEL  256
#define DSTATE  64
#define DCONV   4
#define HEADDIM 16
#define DINNER  512
#define NHEADS  32
#define CONVDIM 640          // DINNER + 2*DSTATE
#define DPROJ   1184         // 2*DINNER + 2*DSTATE + NHEADS
#define DPROJP  1280         // DPROJ padded to 128-col multiple for WMMA tiles
#define NT      (BATCH*SEQ)  // 51200 rows

typedef __attribute__((ext_vector_type(16))) __bf16 v16bf;
typedef __attribute__((ext_vector_type(8)))  float  v8f;

__device__ __forceinline__ float silu_f(float x) {
  return x / (1.0f + __expf(-x));
}

// ---------------------------------------------------------------- fp32 -> bf16
__global__ __launch_bounds__(256) void cvt_bf16_kernel(const float* __restrict__ src,
                                                       __bf16* __restrict__ dst, int n) {
  int i = blockIdx.x * 256 + threadIdx.x;
  if (i < n) dst[i] = (__bf16)src[i];
}

// fp32 [rows,N] -> bf16 [rows,ldp] zero-padded columns
__global__ __launch_bounds__(256) void cvt_pad_bf16_kernel(const float* __restrict__ src,
                                                           __bf16* __restrict__ dst,
                                                           int N, int ldp, int total) {
  int i = blockIdx.x * 256 + threadIdx.x;
  if (i >= total) return;
  int r = i / ldp, c = i % ldp;
  dst[i] = (c < N) ? (__bf16)src[(size_t)r * N + c] : (__bf16)0.0f;
}

// ---------------------------------------------------------------- WMMA GEMM
// C[M,N] f32 = A[M,K] bf16 (row major, ld=K) * B[K,*] bf16 (row major, ld=ldb)
// ldb is a multiple of 128 >= N, so B-tile loads are never guarded.
// Block tile 128x128, K-step 32, 8 waves: wave grid 2(M) x 4(N), each wave
// computes 4x2 v_wmma_f32_16x16x32_bf16 tiles.
// Double-buffered pipeline: tile k+1 is staged (A via CDNA5 async DMA
// global_load_async_to_lds_b128, B via in-flight global loads) while tile k
// is consumed by WMMA; one s_wait_asynccnt + one barrier per K-step.
__device__ __forceinline__ v16bf load_frag_row(const __bf16* rowBase, int kBase) {
  union { v16bf v; uint4 u[2]; } fu;
  fu.u[0] = *(const uint4*)(rowBase + kBase);        // K = kb .. kb+7
  fu.u[1] = *(const uint4*)(rowBase + kBase + 16);   // K = kb+16 .. kb+23
  return fu.v;
}

__global__ __launch_bounds__(256)
void gemm_bf16_kernel(const __bf16* __restrict__ A, const __bf16* __restrict__ B,
                      float* __restrict__ C, int M, int N, int ldb, int K) {
  __shared__ __align__(16) __bf16 As[2][128][40];  // (m, k): 80B row stride
  __shared__ __align__(16) __bf16 Bt[2][128][40];  // (n, k): B tile transposed

  const int bm   = blockIdx.x * 128;
  const int bn   = blockIdx.y * 128;
  const int tid  = threadIdx.x;
  const int lane = tid & 31;
  const int wave = tid >> 5;
  const int wm   = wave & 1;    // 2 wave-rows  -> 64 M each
  const int wn   = wave >> 1;   // 4 wave-cols  -> 32 N each
  const int laneM = lane & 15;
  const int kBase = (lane >> 4) * 8;

  // Per-thread staging coordinates (fixed across K loop)
  const int ar = tid >> 1;               // A row 0..127
  const int ac = (tid & 1) * 16;         // A col 0 / 16
  const unsigned ldsA[2] = { (unsigned)(uintptr_t)&As[0][ar][ac],
                             (unsigned)(uintptr_t)&As[1][ar][ac] };
  unsigned long long gA =
      (unsigned long long)(uintptr_t)(A + (size_t)(bm + ar) * K + ac);

  const int kp = tid >> 4;               // B k-pair 0..15 -> k = 2*kp
  const int nb = (tid & 15) * 8;         // B n base 0..120
  const __bf16* bsrc = B + (size_t)(2 * kp) * ldb + bn + nb;

  v8f acc[4][2] = {};

  // ---------------- prologue: stage tile 0 into buffer 0
  asm volatile("global_load_async_to_lds_b128 %0, %1, off"
               :: "v"(ldsA[0]), "v"(gA) : "memory");
  asm volatile("global_load_async_to_lds_b128 %0, %1, off offset:16"
               :: "v"(ldsA[0]), "v"(gA) : "memory");
  gA += 64;
  {
    uint4 r0 = *(const uint4*)bsrc;
    uint4 r1 = *(const uint4*)(bsrc + ldb);
    const unsigned short* e0 = (const unsigned short*)&r0;
    const unsigned short* e1 = (const unsigned short*)&r1;
#pragma unroll
    for (int i = 0; i < 8; ++i) {
      unsigned pk = (unsigned)e0[i] | ((unsigned)e1[i] << 16);
      *(unsigned*)&Bt[0][nb + i][2 * kp] = pk;
    }
    bsrc += (size_t)32 * ldb;
  }
  asm volatile("s_wait_asynccnt 0" ::: "memory");
  __syncthreads();

  const int ntiles = K >> 5;
  for (int kt = 0; kt < ntiles; ++kt) {
    const int cur = kt & 1;
    const int nxt = cur ^ 1;
    const bool has_next = (kt + 1 < ntiles);

    // ---- stage tile kt+1 (DMA + register loads in flight during compute)
    uint4 r0, r1;
    if (has_next) {
      asm volatile("global_load_async_to_lds_b128 %0, %1, off"
                   :: "v"(ldsA[nxt]), "v"(gA) : "memory");
      asm volatile("global_load_async_to_lds_b128 %0, %1, off offset:16"
                   :: "v"(ldsA[nxt]), "v"(gA) : "memory");
      gA += 64;
      __builtin_prefetch((const void*)(uintptr_t)gA, 0, 3);
      r0 = *(const uint4*)bsrc;
      r1 = *(const uint4*)(bsrc + ldb);
      bsrc += (size_t)32 * ldb;
    }

    // ---- compute on tile kt
    v16bf aF[4], bF[2];
#pragma unroll
    for (int mt = 0; mt < 4; ++mt)
      aF[mt] = load_frag_row(&As[cur][wm * 64 + mt * 16 + laneM][0], kBase);
#pragma unroll
    for (int nt = 0; nt < 2; ++nt)
      bF[nt] = load_frag_row(&Bt[cur][wn * 32 + nt * 16 + laneM][0], kBase);

#pragma unroll
    for (int mt = 0; mt < 4; ++mt)
#pragma unroll
      for (int nt = 0; nt < 2; ++nt)
        acc[mt][nt] = __builtin_amdgcn_wmma_f32_16x16x32_bf16(
            false, aF[mt], false, bF[nt], (short)0, acc[mt][nt], false, false);

    // ---- finish staging tile kt+1: transpose-store B into next buffer
    if (has_next) {
      const unsigned short* e0 = (const unsigned short*)&r0;
      const unsigned short* e1 = (const unsigned short*)&r1;
#pragma unroll
      for (int i = 0; i < 8; ++i) {
        unsigned pk = (unsigned)e0[i] | ((unsigned)e1[i] << 16);
        *(unsigned*)&Bt[nxt][nb + i][2 * kp] = pk;
      }
      asm volatile("s_wait_asynccnt 0" ::: "memory");
    }
    __syncthreads();
  }

  // ---- store: D layout row = 8*(lane>>4)+i, col = lane&15
#pragma unroll
  for (int mt = 0; mt < 4; ++mt)
#pragma unroll
    for (int nt = 0; nt < 2; ++nt) {
      int col = bn + wn * 32 + nt * 16 + laneM;
      if (col < N) {
        int rbase = bm + wm * 64 + mt * 16 + (lane >> 4) * 8;
#pragma unroll
        for (int i = 0; i < 8; ++i)
          C[(size_t)(rbase + i) * N + col] = acc[mt][nt][i];
      }
    }
}

// ---------------------------------------------------------------- depthwise causal conv + SiLU
__global__ __launch_bounds__(256)
void conv_silu_kernel(const float* __restrict__ zx, const float* __restrict__ cw,
                      const float* __restrict__ cb, float* __restrict__ out) {
  int idx = blockIdx.x * 256 + threadIdx.x;              // over NT*CONVDIM
  if (idx >= NT * CONVDIM) return;
  int c  = idx % CONVDIM;
  int bt = idx / CONVDIM;
  int t  = bt % SEQ;
  int b  = bt / SEQ;
  const float* col = zx + (size_t)b * SEQ * DPROJ + DINNER + c;  // xBC section
  float acc = cb[c];
#pragma unroll
  for (int k = 0; k < DCONV; ++k) {
    int tt = t - (DCONV - 1) + k;
    if (tt >= 0) acc = fmaf(cw[c * DCONV + k], col[(size_t)tt * DPROJ], acc);
  }
  out[idx] = acc * (1.0f / (1.0f + __expf(-acc)));
}

// ---------------------------------------------------------------- dt -> softplus, dA
__global__ __launch_bounds__(256)
void dt_kernel(const float* __restrict__ zx, const float* __restrict__ dt_bias,
               const float* __restrict__ A_log, float* __restrict__ dtp,
               float* __restrict__ dAb) {
  int idx = blockIdx.x * 256 + threadIdx.x;              // over NT*NHEADS
  if (idx >= NT * NHEADS) return;
  int h  = idx % NHEADS;
  int bt = idx / NHEADS;
  float d = zx[(size_t)bt * DPROJ + (DINNER + CONVDIM) + h] + dt_bias[h];
  float sp = (d > 20.0f) ? d : log1pf(__expf(d));
  float Ah = -__expf(A_log[h]);
  dtp[idx] = sp;
  dAb[idx] = __expf(sp * Ah);
}

// ---------------------------------------------------------------- selective-state scan
// One wave per (b,h). Lane = (p = lane&15, n-half = lane>>4); 32 state f32 / lane.
__global__ __launch_bounds__(256)
void scan_kernel(const float* __restrict__ conv, const float* __restrict__ dtp,
                 const float* __restrict__ dAb, const float* __restrict__ Dskip,
                 float* __restrict__ y) {
  int wid  = blockIdx.x * 8 + (threadIdx.x >> 5);        // 0 .. 8191
  int lane = threadIdx.x & 31;
  int b = wid >> 5;
  int h = wid & 31;
  int p    = lane & 15;
  int half = lane >> 4;
  float s[32];
#pragma unroll
  for (int j = 0; j < 32; ++j) s[j] = 0.0f;
  const float Dv = Dskip[h];
  for (int t = 0; t < SEQ; ++t) {
    size_t bt = (size_t)b * SEQ + t;
    float dAv = dAb[bt * NHEADS + h];
    float dtv = dtp[bt * NHEADS + h];
    const float* row = conv + bt * CONVDIM;
    float xv  = row[h * HEADDIM + p];
    float dtx = dtv * xv;
    const float* Bp = row + DINNER + half * 32;
    const float* Cp = row + DINNER + DSTATE + half * 32;
    float acc = 0.0f;
#pragma unroll
    for (int j = 0; j < 32; ++j) {
      s[j] = fmaf(dAv, s[j], dtx * Bp[j]);
      acc  = fmaf(s[j], Cp[j], acc);
    }
    acc += __shfl_xor(acc, 16, 32);
    if (half == 0) y[bt * DINNER + h * HEADDIM + p] = acc + Dv * xv;
  }
}

// ---------------------------------------------------------------- gated RMSNorm -> bf16
__global__ __launch_bounds__(256)
void gate_rms_kernel(const float* __restrict__ y, const float* __restrict__ zx,
                     const float* __restrict__ rmsw, __bf16* __restrict__ g) {
  __shared__ float red[256];
  int row = blockIdx.x, tid = threadIdx.x;
  const float* yr = y  + (size_t)row * DINNER;
  const float* zr = zx + (size_t)row * DPROJ;            // z = cols [0,512)
  float v0 = yr[tid]       * silu_f(zr[tid]);
  float v1 = yr[tid + 256] * silu_f(zr[tid + 256]);
  red[tid] = v0 * v0 + v1 * v1;
  __syncthreads();
  for (int sft = 128; sft > 0; sft >>= 1) {
    if (tid < sft) red[tid] += red[tid + sft];
    __syncthreads();
  }
  float scale = rsqrtf(red[0] / (float)DINNER + 1e-5f);
  g[(size_t)row * DINNER + tid]       = (__bf16)(v0 * scale * rmsw[tid]);
  g[(size_t)row * DINNER + tid + 256] = (__bf16)(v1 * scale * rmsw[tid + 256]);
}

// ---------------------------------------------------------------- LayerNorm over 256 cols
// out = LN(a [+ res] [+ bias]); in-place safe (out may equal a).
__global__ __launch_bounds__(256)
void ln256_kernel(const float* __restrict__ a, const float* __restrict__ res,
                  const float* __restrict__ bias, const float* __restrict__ g,
                  const float* __restrict__ be, float* __restrict__ out) {
  __shared__ float red[256];
  int row = blockIdx.x, tid = threadIdx.x;
  float v = a[(size_t)row * DMODEL + tid];
  if (res)  v += res[(size_t)row * DMODEL + tid];
  if (bias) v += bias[tid];
  red[tid] = v;
  __syncthreads();
  for (int s = 128; s > 0; s >>= 1) { if (tid < s) red[tid] += red[tid + s]; __syncthreads(); }
  float mean = red[0] * (1.0f / DMODEL);
  __syncthreads();
  float d = v - mean;
  red[tid] = d * d;
  __syncthreads();
  for (int s = 128; s > 0; s >>= 1) { if (tid < s) red[tid] += red[tid + s]; __syncthreads(); }
  float var = red[0] * (1.0f / DMODEL);
  out[(size_t)row * DMODEL + tid] = d * rsqrtf(var + 1e-12f) * g[tid] + be[tid];
}

// ---------------------------------------------------------------- FFT filter branch (direct DFT)
__global__ __launch_bounds__(128)
void fft_filter_kernel(const float* __restrict__ x, const float* __restrict__ fw,
                       float* __restrict__ xi) {
  __shared__ float xs[SEQ][33];
  __shared__ float Fr[101][33];
  __shared__ float Fi[101][33];
  __shared__ float ct[SEQ];
  __shared__ float st[SEQ];
  int b = blockIdx.x, dg = blockIdx.y, tid = threadIdx.x;

  for (int k = tid; k < SEQ; k += 128) {
    float a = 6.283185307179586f * (float)k / (float)SEQ;
    ct[k] = __cosf(a);
    st[k] = __sinf(a);
  }
  for (int idx = tid; idx < SEQ * 32; idx += 128) {
    int t = idx >> 5, dl = idx & 31;
    xs[t][dl] = x[((size_t)b * SEQ + t) * DMODEL + dg * 32 + dl];
  }
  __syncthreads();

  const float inv = 0.07071067811865475f;                // 1/sqrt(200) (ortho)
  for (int idx = tid; idx < 101 * 32; idx += 128) {
    int f = idx >> 5, dl = idx & 31;
    float re = 0.0f, im = 0.0f;
    for (int t = 0; t < SEQ; ++t) {
      int k = (f * t) % SEQ;
      float v = xs[t][dl];
      re = fmaf(v,  ct[k], re);
      im = fmaf(v, -st[k], im);
    }
    re *= inv; im *= inv;
    int d = dg * 32 + dl;
    float wr = fw[((size_t)f * DMODEL + d) * 2 + 0];
    float wi = fw[((size_t)f * DMODEL + d) * 2 + 1];
    Fr[f][dl] = re * wr - im * wi;
    Fi[f][dl] = re * wi + im * wr;
  }
  __syncthreads();

  for (int idx = tid; idx < SEQ * 32; idx += 128) {
    int t = idx >> 5, dl = idx & 31;
    float acc = 0.0f;
    for (int f = 0; f < 101; ++f) {
      int k = (f * t) % SEQ;
      float wgt = (f == 0 || f == 100) ? 1.0f : 2.0f;
      acc += wgt * (Fr[f][dl] * ct[k] - Fi[f][dl] * st[k]);
    }
    xi[((size_t)b * SEQ + t) * DMODEL + dg * 32 + dl] = acc * inv;
  }
}

// ---------------------------------------------------------------- h = alpha*h1 + beta*h2 (+bf16 copy)
__global__ __launch_bounds__(256)
void combine_kernel(const float* __restrict__ h1, const float* __restrict__ h2,
                    const float* __restrict__ alpha, const float* __restrict__ beta,
                    float* __restrict__ hb, __bf16* __restrict__ hbf) {
  int idx = blockIdx.x * 256 + threadIdx.x;
  if (idx >= NT * DMODEL) return;
  int col = idx & (DMODEL - 1);
  float v = alpha[col] * h1[idx] + beta[col] * h2[idx];
  hb[idx]  = v;
  hbf[idx] = (__bf16)v;
}

// ---------------------------------------------------------------- bias + exact GELU -> bf16
__global__ __launch_bounds__(256)
void bias_gelu_kernel(const float* __restrict__ hid, const float* __restrict__ b1,
                      __bf16* __restrict__ out) {
  int idx = blockIdx.x * 256 + threadIdx.x;
  if (idx >= NT * 1024) return;
  float v = hid[idx] + b1[idx & 1023];
  float gl = 0.5f * v * (1.0f + erff(v * 0.7071067811865475f));
  out[idx] = (__bf16)gl;
}

// ================================================================ launch
extern "C" void kernel_launch(void* const* d_in, const int* in_sizes, int n_in,
                              void* d_out, int out_size, void* d_ws, size_t ws_size,
                              hipStream_t stream) {
  const float* x        = (const float*)d_in[0];
  const float* in_projw = (const float*)d_in[1];
  const float* conv_w   = (const float*)d_in[2];
  const float* conv_b   = (const float*)d_in[3];
  const float* dt_bias  = (const float*)d_in[4];
  const float* A_log    = (const float*)d_in[5];
  const float* D_skip   = (const float*)d_in[6];
  const float* rms_w    = (const float*)d_in[7];
  const float* out_projw= (const float*)d_in[8];
  const float* ln_g     = (const float*)d_in[9];
  const float* ln_b     = (const float*)d_in[10];
  const float* filt_w   = (const float*)d_in[11];
  const float* filt_ln_g= (const float*)d_in[12];
  const float* filt_ln_b= (const float*)d_in[13];
  const float* alpha    = (const float*)d_in[14];
  const float* beta     = (const float*)d_in[15];
  const float* ffn_w1   = (const float*)d_in[16];
  const float* ffn_b1   = (const float*)d_in[17];
  const float* ffn_w2   = (const float*)d_in[18];
  const float* ffn_b2   = (const float*)d_in[19];
  const float* ffn_ln_g = (const float*)d_in[20];
  const float* ffn_ln_b = (const float*)d_in[21];

  char* ws = (char*)d_ws;
  size_t off = 0;
  auto carve = [&](size_t bytes) -> char* {
    char* p = ws + off;
    off = (off + bytes + 255) & ~(size_t)255;
    return p;
  };

  __bf16* xbf   = (__bf16*)carve((size_t)NT * DMODEL * 2);
  __bf16* ipw   = (__bf16*)carve((size_t)DMODEL * DPROJP * 2);  // padded 1280
  __bf16* opw   = (__bf16*)carve((size_t)DINNER * DMODEL * 2);
  __bf16* f1w   = (__bf16*)carve((size_t)DMODEL * 1024 * 2);
  __bf16* f2w   = (__bf16*)carve((size_t)1024 * DMODEL * 2);
  float*  zx    = (float*) carve((size_t)NT * DPROJ * 4);    // later aliased: ffn hidden f32
  float*  convo = (float*) carve((size_t)NT * CONVDIM * 4);  // later aliased: ffn hidden bf16
  float*  dtp   = (float*) carve((size_t)NT * NHEADS * 4);
  float*  dAb   = (float*) carve((size_t)NT * NHEADS * 4);
  float*  yb    = (float*) carve((size_t)NT * DINNER * 4);   // later aliased: ffn out f32
  __bf16* gbf   = (__bf16*)carve((size_t)NT * DINNER * 2);
  float*  h1b   = (float*) carve((size_t)NT * DMODEL * 4);
  float*  h2b   = (float*) carve((size_t)NT * DMODEL * 4);
  float*  hb    = (float*) carve((size_t)NT * DMODEL * 4);
  __bf16* hbf   = (__bf16*)carve((size_t)NT * DMODEL * 2);
  float*  ffnh  = zx;                 // NT*1024 f32 fits in NT*1184 region
  __bf16* ffnhb = (__bf16*)convo;     // NT*1024 bf16 fits in NT*640 f32 region
  float*  ffno  = yb;                 // NT*256 f32 fits in NT*512 region
  (void)ws_size; (void)in_sizes; (void)n_in; (void)out_size;

  auto cvt = [&](const float* s, __bf16* d, int n) {
    cvt_bf16_kernel<<<(n + 255) / 256, 256, 0, stream>>>(s, d, n);
  };
  cvt(x,         xbf, NT * DMODEL);
  cvt_pad_bf16_kernel<<<(DMODEL * DPROJP + 255) / 256, 256, 0, stream>>>(
      in_projw, ipw, DPROJ, DPROJP, DMODEL * DPROJP);
  cvt(out_projw, opw, DINNER * DMODEL);
  cvt(ffn_w1,    f1w, DMODEL * 1024);
  cvt(ffn_w2,    f2w, 1024 * DMODEL);

  // in_proj GEMM: [NT,256] x [256,1184] (B padded to ldb=1280)
  gemm_bf16_kernel<<<dim3(NT / 128, (DPROJ + 127) / 128), 256, 0, stream>>>(
      xbf, ipw, zx, NT, DPROJ, DPROJP, DMODEL);

  conv_silu_kernel<<<(NT * CONVDIM + 255) / 256, 256, 0, stream>>>(zx, conv_w, conv_b, convo);
  dt_kernel<<<(NT * NHEADS + 255) / 256, 256, 0, stream>>>(zx, dt_bias, A_log, dtp, dAb);

  // scan: 8192 (b,h) waves, 8 waves/block
  scan_kernel<<<BATCH * NHEADS / 8, 256, 0, stream>>>(convo, dtp, dAb, D_skip, yb);

  gate_rms_kernel<<<NT, 256, 0, stream>>>(yb, zx, rms_w, gbf);

  // out_proj GEMM: [NT,512] x [512,256]
  gemm_bf16_kernel<<<dim3(NT / 128, DMODEL / 128), 256, 0, stream>>>(
      gbf, opw, h1b, NT, DMODEL, DMODEL, DINNER);
  ln256_kernel<<<NT, 256, 0, stream>>>(h1b, x, nullptr, ln_g, ln_b, h1b);

  // FFT filter branch
  fft_filter_kernel<<<dim3(BATCH, DMODEL / 32), 128, 0, stream>>>(x, filt_w, h2b);
  ln256_kernel<<<NT, 256, 0, stream>>>(h2b, x, nullptr, filt_ln_g, filt_ln_b, h2b);

  combine_kernel<<<(NT * DMODEL + 255) / 256, 256, 0, stream>>>(h1b, h2b, alpha, beta, hb, hbf);

  // FFN GEMM1: [NT,256] x [256,1024]
  gemm_bf16_kernel<<<dim3(NT / 128, 1024 / 128), 256, 0, stream>>>(
      hbf, f1w, ffnh, NT, 1024, 1024, DMODEL);
  bias_gelu_kernel<<<(NT * 1024 + 255) / 256, 256, 0, stream>>>(ffnh, ffn_b1, ffnhb);

  // FFN GEMM2: [NT,1024] x [1024,256]
  gemm_bf16_kernel<<<dim3(NT / 128, DMODEL / 128), 256, 0, stream>>>(
      ffnhb, f2w, ffno, NT, DMODEL, DMODEL, 1024);

  ln256_kernel<<<NT, 256, 0, stream>>>(ffno, hb, ffn_b2, ffn_ln_g, ffn_ln_b, (float*)d_out);
}
